// RewardModelHetero_27384711480155
// MI455X (gfx1250) — matrix operand order repaired and checked by reading, not compiled, not run on hardware
//
#include <hip/hip_runtime.h>
#include <math.h>

typedef _Float16 half_t;
typedef __attribute__((ext_vector_type(16))) _Float16 v16h;
typedef __attribute__((ext_vector_type(8)))  float    v8f;

#define N_NODES  10000
#define N_EDGES  160000
#define N_CROSS  80000
#define N_GRAPHS 8

__device__ __forceinline__ void atomAddF(float* p, float v) {
  __hip_atomic_fetch_add(p, v, __ATOMIC_RELAXED, __HIP_MEMORY_SCOPE_AGENT);
}

// ---------------- utility kernels ----------------
__global__ void zero_f_kernel(float* p, int n) {
  int i = blockIdx.x * blockDim.x + threadIdx.x;
  if (i < n) p[i] = 0.f;
}
__global__ void zero_i_kernel(int* p, int n) {
  int i = blockIdx.x * blockDim.x + threadIdx.x;
  if (i < n) p[i] = 0;
}
__global__ void deg_kernel(const int* __restrict__ src, int* __restrict__ deg, int E) {
  int e = blockIdx.x * blockDim.x + threadIdx.x;
  if (e < E) atomicAdd(&deg[src[e]], 1);
}
__global__ void cnt_node_kernel(const int* __restrict__ batch, int* __restrict__ cnt, int N) {
  int n = blockIdx.x * blockDim.x + threadIdx.x;
  if (n < N) atomicAdd(&cnt[batch[n]], 1);
}
__global__ void cnt_edge_kernel(const int* __restrict__ batch, const int* __restrict__ src,
                                int* __restrict__ cnt, int E) {
  int e = blockIdx.x * blockDim.x + threadIdx.x;
  if (e < E) atomicAdd(&cnt[batch[src[e]]], 1);
}

// Convert (48 x WN) f32 weight matrix into WMMA B-fragment layout f16:
// frag[((t*2+s)*32 + L)*16 + j] = W[K*WN + t*16 + (L&15)], K = s*32 + (L<16?0:16) + j
__global__ void prep_frag_kernel(const float* __restrict__ W, half_t* __restrict__ frag,
                                 int WN, int total) {
  int idx = blockIdx.x * blockDim.x + threadIdx.x;
  if (idx >= total) return;
  int t = idx >> 10;
  int r = idx & 1023;
  int s = r >> 9;
  int L = (r >> 4) & 31;
  int j = r & 15;
  int n = t * 16 + (L & 15);
  int K = s * 32 + ((L < 16) ? 0 : 16) + j;
  frag[idx] = (K < 48) ? (half_t)W[K * WN + n] : (half_t)0.f;
}

// ---------------- node embedding: x0 = mlp64->16->16 ----------------
__global__ void node_emb_kernel(const float* __restrict__ xin,
                                const float* __restrict__ W0, const float* __restrict__ b0,
                                const float* __restrict__ W1, const float* __restrict__ b1,
                                float* __restrict__ x0, int N) {
  int n = blockIdx.x * blockDim.x + threadIdx.x;
  if (n >= N) return;
  float h[16];
#pragma unroll
  for (int i = 0; i < 16; ++i) h[i] = b0[i];
  for (int j = 0; j < 64; ++j) {
    float xv = xin[n * 64 + j];
#pragma unroll
    for (int i = 0; i < 16; ++i) h[i] += xv * W0[j * 16 + i];
  }
#pragma unroll
  for (int i = 0; i < 16; ++i) h[i] = fmaxf(h[i], 0.f);
  float o[16];
#pragma unroll
  for (int i = 0; i < 16; ++i) o[i] = b1[i];
#pragma unroll
  for (int j = 0; j < 16; ++j) {
#pragma unroll
    for (int i = 0; i < 16; ++i) o[i] += h[j] * W1[j * 16 + i];
  }
#pragma unroll
  for (int i = 0; i < 16; ++i) x0[n * 16 + i] = o[i];
}

__global__ void init_x_kernel(const float* __restrict__ x0, float* __restrict__ xb, int N) {
  int idx = blockIdx.x * blockDim.x + threadIdx.x;
  if (idx >= N * 40) return;
  int n = idx / 40, f = idx % 40;
  xb[idx] = (f < 16) ? x0[n * 16 + f] : 0.f;
}

// ---------------- per-edge geometry: smearing -> edge_emb MLP; spherical harmonics ----------------
__global__ void geom_kernel(const float* __restrict__ pos, const int* __restrict__ ei, int E,
                            const float* __restrict__ W0, const float* __restrict__ b0,
                            const float* __restrict__ W1, const float* __restrict__ b1,
                            float* __restrict__ eattr, float* __restrict__ esh) {
  int e = blockIdx.x * blockDim.x + threadIdx.x;
  if (e >= E) return;
  int s = ei[e], d = ei[E + e];
  float v0 = pos[s * 3 + 0] - pos[d * 3 + 0];
  float v1 = pos[s * 3 + 1] - pos[d * 3 + 1];
  float v2 = pos[s * 3 + 2] - pos[d * 3 + 2];
  float dist = sqrtf(v0 * v0 + v1 * v1 + v2 * v2 + 1e-9f);

  const float step = 10.f / 31.f;
  const float coeff = -0.5f / (step * step);
  float h[16];
#pragma unroll
  for (int i = 0; i < 16; ++i) h[i] = b0[i];
  for (int j = 0; j < 32; ++j) {
    float dd = dist - step * (float)j;
    float g = __expf(coeff * dd * dd);
#pragma unroll
    for (int i = 0; i < 16; ++i) h[i] += g * W0[j * 16 + i];
  }
#pragma unroll
  for (int i = 0; i < 16; ++i) h[i] = fmaxf(h[i], 0.f);
  float o[16];
#pragma unroll
  for (int i = 0; i < 16; ++i) o[i] = b1[i];
#pragma unroll
  for (int j = 0; j < 16; ++j) {
#pragma unroll
    for (int i = 0; i < 16; ++i) o[i] += h[j] * W1[j * 16 + i];
  }
#pragma unroll
  for (int i = 0; i < 16; ++i) eattr[e * 16 + i] = o[i];

  float inv = 1.f / dist;
  float nx = v0 * inv, ny = v1 * inv, nz = v2 * inv;
  const float s3 = 1.7320508075688772f;
  const float s15 = 3.872983346207417f;
  const float s5h = 1.118033988749895f;    // sqrt(5)/2
  const float s15h = 1.9364916731037085f;  // sqrt(15)/2
  float* sp = &esh[e * 9];
  sp[0] = 1.f;
  sp[1] = s3 * ny; sp[2] = s3 * nz; sp[3] = s3 * nx;
  sp[4] = s15 * nx * ny;
  sp[5] = s15 * ny * nz;
  sp[6] = s5h * (3.f * nz * nz - 1.f);
  sp[7] = s15 * nx * nz;
  sp[8] = s15h * (nx * nx - ny * ny);
}

// ---------------- fused edge-FC (WMMA) + tensor product + scatter ----------------
template <int LAYER>
__global__ __launch_bounds__(256) void conv_kernel(
    const float* __restrict__ x, const int* __restrict__ src, const int* __restrict__ dst,
    const float* __restrict__ eattr, const float* __restrict__ esh,
    const half_t* __restrict__ w1frag, const half_t* __restrict__ w2frag,
    const float* __restrict__ b1, const float* __restrict__ b2,
    float* __restrict__ sumout, int E) {
  constexpr int WN   = LAYER ? 432 : 320;
  constexpr int NT2  = WN / 16;
  constexpr int OUTD = LAYER ? 40 : 28;

  __shared__ half_t eah[8][16][64];
  __shared__ half_t hh[8][16][64];
  __shared__ float  wt[8][16][16];
  __shared__ float  tpa[8][16][OUTD];

  const int tid = threadIdx.x;
  const int lane = tid & 31;
  const int w = tid >> 5;
  const int ebase = (blockIdx.x * 8 + w) * 16;

  // Stage ea = [edge_attr(16) | x[src,:16] | x[dst,:16]] as f16, K padded to 64.
  for (int idx = lane; idx < 16 * 64; idx += 32) {
    int e = idx >> 6, f = idx & 63;
    int ge = ebase + e;
    float val = 0.f;
    if (ge < E && f < 48) {
      if (f < 16)      val = eattr[ge * 16 + f];
      else if (f < 32) val = x[src[ge] * 40 + (f - 16)];
      else             val = x[dst[ge] * 40 + (f - 32)];
    }
    eah[w][e][f] = (half_t)val;
  }
  asm volatile("s_wait_dscnt 0" ::: "memory");
  __builtin_amdgcn_wave_barrier();

  const int n = lane & 15;
  const int roff = (lane < 16) ? 0 : 8;
  const int kb = (lane < 16) ? 0 : 16;

  // A fragments (K-steps 0..31 and 32..63), reused for all N-tiles of GEMM1.
  v16h a0, a1;
#pragma unroll
  for (int j = 0; j < 16; ++j) {
    int kk = kb + j;
    a0[j] = eah[w][n][kk];
    a1[j] = eah[w][n][32 + kk];
  }

  // GEMM1: h = relu(ea @ W1 + b1), 16x48
  const v16h* f1 = (const v16h*)w1frag;
#pragma unroll
  for (int ct = 0; ct < 3; ++ct) {
    float bias = b1[ct * 16 + n];
    v8f c;
#pragma unroll
    for (int r = 0; r < 8; ++r) c[r] = bias;
    v16h bf0 = f1[(ct * 2 + 0) * 32 + lane];
    v16h bf1 = f1[(ct * 2 + 1) * 32 + lane];
    c = __builtin_amdgcn_wmma_f32_16x16x32_f16(false, a0, false, bf0, (short)0, c, false, false);
    c = __builtin_amdgcn_wmma_f32_16x16x32_f16(false, a1, false, bf1, (short)0, c, false, false);
#pragma unroll
    for (int r = 0; r < 8; ++r) {
      float hv = c[r];
      hv = hv > 0.f ? hv : 0.f;
      hh[w][roff + r][ct * 16 + n] = (half_t)hv;
    }
  }
  for (int idx = lane; idx < 256; idx += 32)  // pad K=48..63 with zeros
    hh[w][idx >> 4][48 + (idx & 15)] = (half_t)0.f;
  asm volatile("s_wait_dscnt 0" ::: "memory");
  __builtin_amdgcn_wave_barrier();

  v16h h0, h1;
#pragma unroll
  for (int j = 0; j < 16; ++j) {
    int kk = kb + j;
    h0[j] = hh[w][n][kk];
    h1[j] = hh[w][n][32 + kk];
  }

  // Per-lane TP precompute (lane e owns edge ebase+e; lanes 16..31 compute harmless dupes).
  int eg = ebase + lane;
  int eld = (eg < E) ? eg : (E - 1);
  int dn = dst[eld];
  float sh[9];
#pragma unroll
  for (int i = 0; i < 9; ++i) sh[i] = esh[eld * 9 + i];
  float xs[16];
#pragma unroll
  for (int i = 0; i < 16; ++i) xs[i] = x[dn * 40 + i];

  float vv[4][3], dvs[4], cxp[4][3], tq[4][3];
  if (LAYER == 1) {
    const float ca = 0.31622776601683794f;  // sqrt(0.1)
    const float cb = 0.18257418583505536f;  // sqrt(1/30)
    const float cc = 0.3651483716701107f;   // 2*cb
#pragma unroll
    for (int u = 0; u < 4; ++u) {
#pragma unroll
      for (int k = 0; k < 3; ++k) vv[u][k] = x[dn * 40 + 16 + u * 3 + k];
      dvs[u] = vv[u][0] * sh[1] + vv[u][1] * sh[2] + vv[u][2] * sh[3];
      // cross product in e3nn (y,z,x) order
      cxp[u][0] = vv[u][1] * sh[3] - vv[u][2] * sh[2];
      cxp[u][1] = vv[u][2] * sh[1] - vv[u][0] * sh[3];
      cxp[u][2] = vv[u][0] * sh[2] - vv[u][1] * sh[1];
      // real CG 1x2->1 (normalized)
      float s0 = sh[4], s1v = sh[5], s2v = sh[6], s3v = sh[7], s4v = sh[8];
      tq[u][0] = ca * (vv[u][2] * s0 + vv[u][1] * s1v) - cb * vv[u][0] * s2v - ca * vv[u][0] * s4v;
      tq[u][1] = ca * (vv[u][2] * s3v + vv[u][0] * s1v) + cc * vv[u][1] * s2v;
      tq[u][2] = ca * (vv[u][0] * s0 + vv[u][1] * s3v) - cb * vv[u][2] * s2v + ca * vv[u][2] * s4v;
    }
  }

  for (int idx = lane; idx < 16 * OUTD; idx += 32) tpa[w][idx / OUTD][idx % OUTD] = 0.f;
  asm volatile("s_wait_dscnt 0" ::: "memory");
  __builtin_amdgcn_wave_barrier();

  // GEMM2 tile-by-tile, each 16-column weight tile consumed immediately by the TP.
  const v16h* f2 = (const v16h*)w2frag;
  for (int t = 0; t < NT2; ++t) {
    float bias = b2[t * 16 + n];
    v8f c;
#pragma unroll
    for (int r = 0; r < 8; ++r) c[r] = bias;
    v16h bf0 = f2[(t * 2 + 0) * 32 + lane];
    v16h bf1 = f2[(t * 2 + 1) * 32 + lane];
    c = __builtin_amdgcn_wmma_f32_16x16x32_f16(false, h0, false, bf0, (short)0, c, false, false);
    c = __builtin_amdgcn_wmma_f32_16x16x32_f16(false, h1, false, bf1, (short)0, c, false, false);
#pragma unroll
    for (int r = 0; r < 8; ++r) wt[w][roff + r][n] = c[r];
    asm volatile("s_wait_dscnt 0" ::: "memory");
    __builtin_amdgcn_wave_barrier();

    if (lane < 16) {
      float* acc = &tpa[w][lane][0];
#pragma unroll
      for (int ci = 0; ci < 16; ++ci) {
        float wvv = wt[w][lane][ci];
        int gc = t * 16 + ci;
        if (LAYER == 0) {
          if (gc < 256) {  // (0x0->0): 16x16 weights
            int u = gc >> 4, ww2 = gc & 15;
            acc[ww2] += 0.25f * xs[u] * wvv;
          } else {         // (0x1->1): 16x4
            int q = gc - 256; int u = q >> 2, ww2 = q & 3;
            float bv = 0.25f * xs[u] * wvv;
            acc[16 + 3 * ww2 + 0] += bv * sh[1];
            acc[16 + 3 * ww2 + 1] += bv * sh[2];
            acc[16 + 3 * ww2 + 2] += bv * sh[3];
          }
        } else {
          if (gc < 256) {        // P0 (0x0->0): 16x16, alpha=1/sqrt(20)
            int u = gc >> 4, ww2 = gc & 15;
            acc[ww2] += 0.22360679774997896f * xs[u] * wvv;
          } else if (gc < 320) { // P1 (0x1->1): 16x4, 1/sqrt(24)
            int q = gc - 256; int u = q >> 2, ww2 = q & 3;
            float bv = 0.2041241452319315f * xs[u] * wvv;
            acc[16 + 3 * ww2 + 0] += bv * sh[1];
            acc[16 + 3 * ww2 + 1] += bv * sh[2];
            acc[16 + 3 * ww2 + 2] += bv * sh[3];
          } else if (gc < 336) { // P2 (1x0->1): 4x4, 1/sqrt(24)
            int q = gc - 320; int u = q >> 2, ww2 = q & 3;
            float a1c = 0.2041241452319315f * wvv;
            acc[16 + 3 * ww2 + 0] += a1c * vv[u][0];
            acc[16 + 3 * ww2 + 1] += a1c * vv[u][1];
            acc[16 + 3 * ww2 + 2] += a1c * vv[u][2];
          } else if (gc < 400) { // P3 (1x1->0): 4x16, 1/sqrt(60)
            int q = gc - 336; int u = q >> 4, ww2 = q & 15;
            acc[ww2] += 0.12909944487358055f * dvs[u] * wvv;
          } else if (gc < 416) { // P4 (1x1->1 cross): 4x4, 1/sqrt(8)
            int q = gc - 400; int u = q >> 2, ww2 = q & 3;
            float a4c = 0.3535533905932738f * wvv;
            acc[28 + 3 * ww2 + 0] += a4c * cxp[u][0];
            acc[28 + 3 * ww2 + 1] += a4c * cxp[u][1];
            acc[28 + 3 * ww2 + 2] += a4c * cxp[u][2];
          } else {               // P5 (1x2->1): 4x4, 1/sqrt(8)
            int q = gc - 416; int u = q >> 2, ww2 = q & 3;
            float a5c = 0.3535533905932738f * wvv;
            acc[16 + 3 * ww2 + 0] += a5c * tq[u][0];
            acc[16 + 3 * ww2 + 1] += a5c * tq[u][1];
            acc[16 + 3 * ww2 + 2] += a5c * tq[u][2];
          }
        }
      }
    }
    __builtin_amdgcn_wave_barrier();
  }
  asm volatile("s_wait_dscnt 0" ::: "memory");
  __builtin_amdgcn_wave_barrier();

  // Scatter-add per-edge TP output into per-src-node sum (segment mean numerator).
  for (int idx = lane; idx < 16 * OUTD; idx += 32) {
    int e2 = idx / OUTD, f = idx % OUTD;
    int ge = ebase + e2;
    if (ge < E) atomAddF(&sumout[src[ge] * OUTD + f], tpa[w][e2][f]);
  }
}

// ---------------- x update: pad(x) + sum_e/deg_e + sum_c/deg_c ----------------
__global__ void update_kernel(const float* __restrict__ xold, float* __restrict__ xnew,
                              const float* __restrict__ sum_e, const float* __restrict__ sum_c,
                              const int* __restrict__ deg_e, const int* __restrict__ deg_c,
                              int N, int IND, int OUTD) {
  int idx = blockIdx.x * blockDim.x + threadIdx.x;
  if (idx >= N * 40) return;
  int n = idx / 40, f = idx % 40;
  float val = 0.f;
  if (f < OUTD) {
    float a = (f < IND) ? xold[n * 40 + f] : 0.f;
    float de = (float)(deg_e[n] > 0 ? deg_e[n] : 1);
    float dc = (float)(deg_c[n] > 0 ? deg_c[n] : 1);
    val = a + sum_e[n * OUTD + f] / de + sum_c[n * OUTD + f] / dc;
  }
  xnew[idx] = val;
}

// ---------------- energy heads ----------------
__global__ void eedges_kernel(const float* __restrict__ xf, const float* __restrict__ eattr,
                              const int* __restrict__ src, const int* __restrict__ dst,
                              const int* __restrict__ batch,
                              const float* __restrict__ W0, const float* __restrict__ b0,
                              const float* __restrict__ W1, const float* __restrict__ b1,
                              const float* __restrict__ W2, const float* __restrict__ b2,
                              float* __restrict__ gsum, int E) {
  int e = blockIdx.x * blockDim.x + threadIdx.x;
  if (e >= E) return;
  int s = src[e], d = dst[e];
  float in[48];
#pragma unroll
  for (int i = 0; i < 16; ++i) {
    in[i] = eattr[e * 16 + i];
    in[16 + i] = xf[s * 40 + i];
    in[32 + i] = xf[d * 40 + i];
  }
  float h1[16];
#pragma unroll
  for (int i = 0; i < 16; ++i) h1[i] = b0[i];
  for (int j = 0; j < 48; ++j) {
#pragma unroll
    for (int i = 0; i < 16; ++i) h1[i] += in[j] * W0[j * 16 + i];
  }
#pragma unroll
  for (int i = 0; i < 16; ++i) h1[i] = fmaxf(h1[i], 0.f);
  float h2[16];
#pragma unroll
  for (int i = 0; i < 16; ++i) h2[i] = b1[i];
#pragma unroll
  for (int j = 0; j < 16; ++j) {
#pragma unroll
    for (int i = 0; i < 16; ++i) h2[i] += h1[j] * W1[j * 16 + i];
  }
  float val = b2[0];
#pragma unroll
  for (int j = 0; j < 16; ++j) val += fmaxf(h2[j], 0.f) * W2[j];
  atomAddF(&gsum[batch[s]], val);
}

__global__ void enodes_kernel(const float* __restrict__ xf, const int* __restrict__ batch,
                              const float* __restrict__ W0, const float* __restrict__ b0,
                              const float* __restrict__ W1, const float* __restrict__ b1,
                              const float* __restrict__ W2, const float* __restrict__ b2,
                              float* __restrict__ gsum, int N) {
  int n = blockIdx.x * blockDim.x + threadIdx.x;
  if (n >= N) return;
  float h1[32];
#pragma unroll
  for (int i = 0; i < 32; ++i) h1[i] = b0[i];
#pragma unroll
  for (int j = 0; j < 16; ++j) {
    float xv = xf[n * 40 + j];
#pragma unroll
    for (int i = 0; i < 32; ++i) h1[i] += xv * W0[j * 32 + i];
  }
#pragma unroll
  for (int i = 0; i < 32; ++i) h1[i] = fmaxf(h1[i], 0.f);
  float h2[16];
#pragma unroll
  for (int i = 0; i < 16; ++i) h2[i] = b1[i];
#pragma unroll
  for (int j = 0; j < 32; ++j) {
#pragma unroll
    for (int i = 0; i < 16; ++i) h2[i] += h1[j] * W1[j * 16 + i];
  }
  float val = b2[0];
#pragma unroll
  for (int j = 0; j < 16; ++j) val += fmaxf(h2[j], 0.f) * W2[j];
  atomAddF(&gsum[batch[n]], val);
}

__global__ void graph_energy_kernel(const float* __restrict__ gse, const float* __restrict__ gsn,
                                    const int* __restrict__ cnte, const int* __restrict__ cntn,
                                    float* __restrict__ eres) {
  int g = threadIdx.x;
  if (g < N_GRAPHS) {
    float ce = (float)(cnte[g] > 0 ? cnte[g] : 1);
    float cn = (float)(cntn[g] > 0 ? cntn[g] : 1);
    eres[g] = gse[g] / ce + gsn[g] / cn;
  }
}

__global__ void emit_kernel(const float* __restrict__ eres, float* __restrict__ out) {
  int g = threadIdx.x;
  if (g < N_GRAPHS) {
    out[g] = eres[g];
    out[8 + g] = eres[8 + g];
    out[16 + g] = eres[g] - eres[8 + g];
  }
}

// ---------------- host driver ----------------
extern "C" void kernel_launch(void* const* d_in, const int* in_sizes, int n_in,
                              void* d_out, int out_size, void* d_ws, size_t ws_size,
                              hipStream_t stream) {
  (void)in_sizes; (void)n_in; (void)out_size; (void)ws_size;
  const float* x_in = (const float*)d_in[0];
  const float* pos = (const float*)d_in[1];
  const float* posr = (const float*)d_in[2];
  const int* eidx = (const int*)d_in[3];
  const int* cidx = (const int*)d_in[4];
  const int* rcidx = (const int*)d_in[5];
  const int* batch = (const int*)d_in[6];
  // params flattened as JAX pytree leaves (dict keys sorted)
  const float* convW1[2] = {(const float*)d_in[7], (const float*)d_in[11]};
  const float* convB1[2] = {(const float*)d_in[8], (const float*)d_in[12]};
  const float* convW2[2] = {(const float*)d_in[9], (const float*)d_in[13]};
  const float* convB2[2] = {(const float*)d_in[10], (const float*)d_in[14]};
  const float* xconvW1[2] = {(const float*)d_in[15], (const float*)d_in[19]};
  const float* xconvB1[2] = {(const float*)d_in[16], (const float*)d_in[20]};
  const float* xconvW2[2] = {(const float*)d_in[17], (const float*)d_in[21]};
  const float* xconvB2[2] = {(const float*)d_in[18], (const float*)d_in[22]};
  const float* cembW0 = (const float*)d_in[23];
  const float* cembB0 = (const float*)d_in[24];
  const float* cembW1 = (const float*)d_in[25];
  const float* cembB1 = (const float*)d_in[26];
  const float* eeW0 = (const float*)d_in[27];
  const float* eeB0 = (const float*)d_in[28];
  const float* eeW1 = (const float*)d_in[29];
  const float* eeB1 = (const float*)d_in[30];
  const float* eeW2 = (const float*)d_in[31];
  const float* eeB2 = (const float*)d_in[32];
  const float* enW0 = (const float*)d_in[33];
  const float* enB0 = (const float*)d_in[34];
  const float* enW1 = (const float*)d_in[35];
  const float* enB1 = (const float*)d_in[36];
  const float* enW2 = (const float*)d_in[37];
  const float* enB2 = (const float*)d_in[38];
  const float* embW0 = (const float*)d_in[39];
  const float* embB0 = (const float*)d_in[40];
  const float* embW1 = (const float*)d_in[41];
  const float* embB1 = (const float*)d_in[42];
  const float* nW0 = (const float*)d_in[43];
  const float* nB0 = (const float*)d_in[44];
  const float* nW1 = (const float*)d_in[45];
  const float* nB1 = (const float*)d_in[46];

  char* ws = (char*)d_ws;
  size_t off = 0;
  auto alloc = [&](size_t bytes) -> void* {
    void* p = ws + off;
    off = (off + bytes + 255) & ~(size_t)255;
    return p;
  };

  float* x0 = (float*)alloc((size_t)N_NODES * 16 * 4);
  float* xA = (float*)alloc((size_t)N_NODES * 40 * 4);
  float* xB = (float*)alloc((size_t)N_NODES * 40 * 4);
  float* eaE = (float*)alloc((size_t)N_EDGES * 16 * 4);
  float* shE = (float*)alloc((size_t)N_EDGES * 9 * 4);
  float* eaC = (float*)alloc((size_t)N_CROSS * 16 * 4);
  float* shC = (float*)alloc((size_t)N_CROSS * 9 * 4);
  float* sumE = (float*)alloc((size_t)N_NODES * 40 * 4);
  float* sumC = (float*)alloc((size_t)N_NODES * 40 * 4);
  int* degE = (int*)alloc((size_t)N_NODES * 4);
  int* degC = (int*)alloc((size_t)N_NODES * 4);
  int* cntN = (int*)alloc(64);
  int* cntE = (int*)alloc(64);
  float* gse = (float*)alloc(64);
  float* gsn = (float*)alloc(64);
  float* eres = (float*)alloc(64);
  const int wnum[2] = {320, 432};
  half_t* f1c[2]; half_t* f2c[2]; half_t* f1x[2]; half_t* f2x[2];
  for (int i = 0; i < 2; ++i) {
    f1c[i] = (half_t*)alloc(3072 * 2);
    f2c[i] = (half_t*)alloc((size_t)(wnum[i] / 16) * 1024 * 2);
    f1x[i] = (half_t*)alloc(3072 * 2);
    f2x[i] = (half_t*)alloc((size_t)(wnum[i] / 16) * 1024 * 2);
  }

  auto grid = [](int n, int bs) { return (n + bs - 1) / bs; };

  // weight fragment prep (f16, WMMA B layout)
  for (int i = 0; i < 2; ++i) {
    int tot1 = 3 * 1024;
    int tot2 = (wnum[i] / 16) * 1024;
    prep_frag_kernel<<<grid(tot1, 256), 256, 0, stream>>>(convW1[i], f1c[i], 48, tot1);
    prep_frag_kernel<<<grid(tot2, 256), 256, 0, stream>>>(convW2[i], f2c[i], wnum[i], tot2);
    prep_frag_kernel<<<grid(tot1, 256), 256, 0, stream>>>(xconvW1[i], f1x[i], 48, tot1);
    prep_frag_kernel<<<grid(tot2, 256), 256, 0, stream>>>(xconvW2[i], f2x[i], wnum[i], tot2);
  }

  node_emb_kernel<<<grid(N_NODES, 128), 128, 0, stream>>>(x_in, nW0, nB0, nW1, nB1, x0, N_NODES);

  zero_i_kernel<<<grid(N_NODES, 256), 256, 0, stream>>>(degE, N_NODES);
  deg_kernel<<<grid(N_EDGES, 256), 256, 0, stream>>>(eidx, degE, N_EDGES);
  zero_i_kernel<<<1, 64, 0, stream>>>(cntN, 16);
  zero_i_kernel<<<1, 64, 0, stream>>>(cntE, 16);
  cnt_node_kernel<<<grid(N_NODES, 256), 256, 0, stream>>>(batch, cntN, N_NODES);
  cnt_edge_kernel<<<grid(N_EDGES, 256), 256, 0, stream>>>(batch, eidx, cntE, N_EDGES);

  for (int en = 0; en < 2; ++en) {
    const float* P = en ? posr : pos;
    const int* CI = en ? rcidx : cidx;

    geom_kernel<<<grid(N_EDGES, 128), 128, 0, stream>>>(P, eidx, N_EDGES, embW0, embB0, embW1, embB1, eaE, shE);
    geom_kernel<<<grid(N_CROSS, 128), 128, 0, stream>>>(P, CI, N_CROSS, cembW0, cembB0, cembW1, cembB1, eaC, shC);
    zero_i_kernel<<<grid(N_NODES, 256), 256, 0, stream>>>(degC, N_NODES);
    deg_kernel<<<grid(N_CROSS, 256), 256, 0, stream>>>(CI, degC, N_CROSS);
    init_x_kernel<<<grid(N_NODES * 40, 256), 256, 0, stream>>>(x0, xA, N_NODES);

    // layer 0
    zero_f_kernel<<<grid(N_NODES * 40, 256), 256, 0, stream>>>(sumE, N_NODES * 40);
    zero_f_kernel<<<grid(N_NODES * 40, 256), 256, 0, stream>>>(sumC, N_NODES * 40);
    conv_kernel<0><<<grid(N_EDGES, 128), 256, 0, stream>>>(xA, eidx, eidx + N_EDGES, eaE, shE,
                                                           f1c[0], f2c[0], convB1[0], convB2[0], sumE, N_EDGES);
    conv_kernel<0><<<grid(N_CROSS, 128), 256, 0, stream>>>(xA, CI, CI + N_CROSS, eaC, shC,
                                                           f1x[0], f2x[0], xconvB1[0], xconvB2[0], sumC, N_CROSS);
    update_kernel<<<grid(N_NODES * 40, 256), 256, 0, stream>>>(xA, xB, sumE, sumC, degE, degC, N_NODES, 16, 28);

    // layer 1
    zero_f_kernel<<<grid(N_NODES * 40, 256), 256, 0, stream>>>(sumE, N_NODES * 40);
    zero_f_kernel<<<grid(N_NODES * 40, 256), 256, 0, stream>>>(sumC, N_NODES * 40);
    conv_kernel<1><<<grid(N_EDGES, 128), 256, 0, stream>>>(xB, eidx, eidx + N_EDGES, eaE, shE,
                                                           f1c[1], f2c[1], convB1[1], convB2[1], sumE, N_EDGES);
    conv_kernel<1><<<grid(N_CROSS, 128), 256, 0, stream>>>(xB, CI, CI + N_CROSS, eaC, shC,
                                                           f1x[1], f2x[1], xconvB1[1], xconvB2[1], sumC, N_CROSS);
    update_kernel<<<grid(N_NODES * 40, 256), 256, 0, stream>>>(xB, xA, sumE, sumC, degE, degC, N_NODES, 28, 40);

    // energy heads
    zero_f_kernel<<<1, 64, 0, stream>>>(gse, 16);
    zero_f_kernel<<<1, 64, 0, stream>>>(gsn, 16);
    eedges_kernel<<<grid(N_EDGES, 128), 128, 0, stream>>>(xA, eaE, eidx, eidx + N_EDGES, batch,
                                                          eeW0, eeB0, eeW1, eeB1, eeW2, eeB2, gse, N_EDGES);
    enodes_kernel<<<grid(N_NODES, 128), 128, 0, stream>>>(xA, batch, enW0, enB0, enW1, enB1, enW2, enB2, gsn, N_NODES);
    graph_energy_kernel<<<1, 32, 0, stream>>>(gse, gsn, cntE, cntN, eres + en * 8);
  }

  emit_kernel<<<1, 32, 0, stream>>>(eres, (float*)d_out);
}
